// AdditiveAttention_1056561955202
// MI455X (gfx1250) — matrix operand order repaired and checked by reading, not compiled, not run on hardware
//
#include <hip/hip_runtime.h>
#include <hip/hip_bf16.h>

#define B_ 32
#define T_ 2048
#define D_ 512
#define U_ 512

typedef __attribute__((ext_vector_type(16))) __bf16 v16bf;
typedef __attribute__((ext_vector_type(8)))  __bf16 v8bf;
typedef __attribute__((ext_vector_type(8)))  float  v8f;
typedef __attribute__((ext_vector_type(4)))  float  v4f;

union BF16x16 { v16bf v; v8bf h[2]; };

// LDS row stride: 512 + 8 pad bf16 -> consecutive rows land 4 banks apart;
// 16-lane b128 fragment reads are conflict-free.
#define AS 520

// ---------------------------------------------------------------------------
// Pass 0: q[b][u] = values[b,T-1,:] @ W2[:,u] + W2_b[u] + W1_b[u]
// ---------------------------------------------------------------------------
__global__ __launch_bounds__(256) void query_kernel(
    const float* __restrict__ values, const float* __restrict__ W2,
    const float* __restrict__ W2b, const float* __restrict__ W1b,
    float* __restrict__ qws) {
  int idx = blockIdx.x * 256 + threadIdx.x;   // [0, B_*U_)
  int b = idx >> 9;
  int u = idx & (U_ - 1);
  const float* last = values + ((size_t)b * T_ + (T_ - 1)) * D_;
  float acc = W2b[u] + W1b[u];
  for (int k = 0; k < D_; ++k) acc += last[k] * W2[(size_t)k * U_ + u];
  qws[idx] = acc;
}

// ---------------------------------------------------------------------------
// Pass 1: fused keys-GEMM (bf16 WMMA) + tanh + V-dot -> scores[b][t]
// grid: (T_/64, B_), block: 256 (8 waves)
// ---------------------------------------------------------------------------
__global__ __launch_bounds__(256) void score_kernel(
    const float* __restrict__ values, const float* __restrict__ W1,
    const float* __restrict__ Vw, const float* __restrict__ Vb,
    const float* __restrict__ qws, float* __restrict__ scores) {
  __shared__ __bf16 Asm[64 * AS];      // values tile  [64 rows t][512 k]
  __shared__ __bf16 Bsm[64 * AS];      // W1^T chunk   [64 rows u][512 k]
  __shared__ float  score_lds[64];

  const int tid = threadIdx.x;
  const int b   = blockIdx.y;
  const int t0  = blockIdx.x * 64;

  const int wave = tid >> 5;
  const int lane = tid & 31;
  const int g    = lane >> 4;          // half-wave group
  const int ln   = lane & 15;
  const int m0   = (wave & 3) * 16;    // row tile (t) owned by this wave
  const int c0   = (wave >> 2) * 32;   // 2 col tiles (u) owned by this wave

  // Load A tile once: 64x512 f32 -> bf16, 8 elements per thread per step
  // (two global b128 loads -> packed cvt -> one ds_store_b128)
  for (int idx = tid; idx < 64 * 64; idx += 256) {
    int k8 = idx & 63;                 // vec8 index along k (coalesced)
    int m  = idx >> 6;
    int k  = k8 * 8;
    const v4f* gp = (const v4f*)(values + ((size_t)b * T_ + t0 + m) * D_ + k);
    v4f x = gp[0], y = gp[1];
    v8bf o;
    o[0] = (__bf16)x[0]; o[1] = (__bf16)x[1]; o[2] = (__bf16)x[2]; o[3] = (__bf16)x[3];
    o[4] = (__bf16)y[0]; o[5] = (__bf16)y[1]; o[6] = (__bf16)y[2]; o[7] = (__bf16)y[3];
    *(v8bf*)&Asm[m * AS + k] = o;
  }
  if (tid < 64) score_lds[tid] = 0.0f;
  __syncthreads();

  for (int uc = 0; uc < 8; ++uc) {
    const int u0 = uc * 64;
    // Load W1^T chunk: Bsm[u][k] = W1[k][u0+u]; adjacent lanes -> adjacent u
    // (global coalesced per k), 8 k's packed into one ds_store_b128.
    for (int idx = tid; idx < 64 * 64; idx += 256) {
      int u  = idx & 63;
      int k  = (idx >> 6) * 8;
      const float* gp = W1 + (size_t)k * U_ + u0 + u;
      v8bf o;
#pragma unroll
      for (int j = 0; j < 8; ++j) o[j] = (__bf16)gp[(size_t)j * U_];
      *(v8bf*)&Bsm[u * AS + k] = o;
    }
    __syncthreads();

    // per-lane query / V weights for this wave's two column tiles
    const float q0  = qws[b * U_ + u0 + c0 + ln];
    const float q1  = qws[b * U_ + u0 + c0 + 16 + ln];
    const float vw0 = Vw[u0 + c0 + ln];
    const float vw1 = Vw[u0 + c0 + 16 + ln];

    v8f acc0 = {}, acc1 = {};
    const __bf16* arow  = &Asm[(m0 + ln) * AS];
    const __bf16* brow0 = &Bsm[(c0 + ln) * AS];
    const __bf16* brow1 = &Bsm[(c0 + 16 + ln) * AS];

#pragma unroll
    for (int k0 = 0; k0 < 512; k0 += 32) {
      // A fragment: elems i = K g*16 + i of row ln (contiguous)
      BF16x16 af;
      af.h[0] = *(const v8bf*)(arow + k0 + g * 16);
      af.h[1] = *(const v8bf*)(arow + k0 + g * 16 + 8);
      // B fragments: elems i = K g*16 + i of column ln (contiguous)
      BF16x16 bf0, bf1;
      bf0.h[0] = *(const v8bf*)(brow0 + k0 + g * 16);
      bf0.h[1] = *(const v8bf*)(brow0 + k0 + g * 16 + 8);
      bf1.h[0] = *(const v8bf*)(brow1 + k0 + g * 16);
      bf1.h[1] = *(const v8bf*)(brow1 + k0 + g * 16 + 8);

      acc0 = __builtin_amdgcn_wmma_f32_16x16x32_bf16(
          false, af.v, false, bf0.v, (short)0, acc0, false, false);
      acc1 = __builtin_amdgcn_wmma_f32_16x16x32_bf16(
          false, af.v, false, bf1.v, (short)0, acc1, false, false);
    }

    // Epilogue: score partial = sum_u tanh(keys + q) * Vw
    // C layout: acc[r] = D[m0 + r + 8*g][ln]
#pragma unroll
    for (int r = 0; r < 8; ++r) {
      float s = tanhf(acc0[r] + q0) * vw0 + tanhf(acc1[r] + q1) * vw1;
      s += __shfl_xor(s, 8, 32);
      s += __shfl_xor(s, 4, 32);
      s += __shfl_xor(s, 2, 32);
      s += __shfl_xor(s, 1, 32);
      if (ln == 0) atomicAdd(&score_lds[m0 + r + 8 * g], s);  // ds_add_f32
    }
    __syncthreads();   // protect Bsm reload & score_lds
  }

  if (tid < 64) scores[(size_t)b * T_ + t0 + tid] = score_lds[tid] + Vb[0];
}

// ---------------------------------------------------------------------------
// Pass 2a: per-batch softmax stats (max, sumexp)
// ---------------------------------------------------------------------------
__global__ __launch_bounds__(256) void softmax_stats_kernel(
    const float* __restrict__ scores, float* __restrict__ mxs) {
  __shared__ float red[256];
  int b = blockIdx.x, tid = threadIdx.x;
  float m = -3.4e38f;
  for (int t = tid; t < T_; t += 256) m = fmaxf(m, scores[(size_t)b * T_ + t]);
  red[tid] = m; __syncthreads();
  for (int s = 128; s > 0; s >>= 1) {
    if (tid < s) red[tid] = fmaxf(red[tid], red[tid + s]);
    __syncthreads();
  }
  float mx = red[0]; __syncthreads();
  float sum = 0.0f;
  for (int t = tid; t < T_; t += 256) sum += __expf(scores[(size_t)b * T_ + t] - mx);
  red[tid] = sum; __syncthreads();
  for (int s = 128; s > 0; s >>= 1) {
    if (tid < s) red[tid] += red[tid + s];
    __syncthreads();
  }
  if (tid == 0) { mxs[b] = mx; mxs[B_ + b] = red[0]; }
}

// ---------------------------------------------------------------------------
// Pass 2b: partial weighted sums over T chunks (deterministic, no atomics)
// grid: (T_/128, B_), block 256 (each thread owns two consecutive d)
// ---------------------------------------------------------------------------
__global__ __launch_bounds__(256) void weighted_partial_kernel(
    const float* __restrict__ values, const float* __restrict__ scores,
    const float* __restrict__ mxs, float* __restrict__ parts) {
  int b = blockIdx.y, ch = blockIdx.x, tid = threadIdx.x;
  float mx = mxs[b];
  float inv = 1.0f / mxs[B_ + b];
  int t0 = ch * 128;
  float a0 = 0.0f, a1 = 0.0f;
  for (int i = 0; i < 128; ++i) {
    int t = t0 + i;
    float w = __expf(scores[(size_t)b * T_ + t] - mx) * inv;
    const float2* vp = (const float2*)(values + ((size_t)b * T_ + t) * D_);
    float2 v = vp[tid];
    a0 += w * v.x;
    a1 += w * v.y;
  }
  float2* pp = (float2*)(parts + ((size_t)ch * B_ + b) * D_);
  pp[tid] = make_float2(a0, a1);
}

// ---------------------------------------------------------------------------
// Pass 2c: reduce the 16 chunk partials -> out[B][D]
// ---------------------------------------------------------------------------
__global__ __launch_bounds__(256) void finalize_kernel(
    const float* __restrict__ parts, float* __restrict__ out) {
  int i = blockIdx.x * 256 + threadIdx.x;   // [0, B_*D_)
  float s = 0.0f;
#pragma unroll
  for (int c = 0; c < 16; ++c) s += parts[(size_t)c * B_ * D_ + i];
  out[i] = s;
}

extern "C" void kernel_launch(void* const* d_in, const int* in_sizes, int n_in,
                              void* d_out, int out_size, void* d_ws, size_t ws_size,
                              hipStream_t stream) {
  const float* values = (const float*)d_in[0];
  const float* W1w    = (const float*)d_in[1];
  const float* W1b    = (const float*)d_in[2];
  const float* W2w    = (const float*)d_in[3];
  const float* W2b    = (const float*)d_in[4];
  const float* Vw     = (const float*)d_in[5];
  const float* Vb     = (const float*)d_in[6];
  float* out = (float*)d_out;

  // workspace layout (floats): q[B*U] | scores[B*T] | stats[2*B] | parts[16*B*D]
  float* qws    = (float*)d_ws;
  float* scores = qws + (size_t)B_ * U_;
  float* mxs    = scores + (size_t)B_ * T_;
  float* parts  = mxs + 2 * B_;

  query_kernel<<<(B_ * U_) / 256, 256, 0, stream>>>(values, W2w, W2b, W1b, qws);
  score_kernel<<<dim3(T_ / 64, B_), 256, 0, stream>>>(values, W1w, Vw, Vb, qws, scores);
  softmax_stats_kernel<<<B_, 256, 0, stream>>>(scores, mxs);
  weighted_partial_kernel<<<dim3(T_ / 128, B_), 256, 0, stream>>>(values, scores, mxs, parts);
  finalize_kernel<<<(B_ * D_) / 256, 256, 0, stream>>>(parts, out);
}
